// SparseResBlock3D_58007828300210
// MI455X (gfx1250) — compile-verified, matrix operand order checked by reading, not run on hardware
//
#include <hip/hip_runtime.h>
#include <hip/hip_bf16.h>

typedef __attribute__((ext_vector_type(16))) _Float16 v16h;
typedef __attribute__((ext_vector_type(8)))  _Float16 v8h;
typedef __attribute__((ext_vector_type(8)))  float    v8f;

#define CCH 27           // taps
#define NF  (27*4*8)     // fragments per conv: tap * kchunk * ntile
#define WFRAG_HALFS (NF*512)
#define MT  32           // rows per block (two 16-row M-tiles)

__device__ __forceinline__ float silu_f(float x) {
    return x / (1.0f + __expf(-x));
}

// ---------------------------------------------------------------------------
// Embedding MLP: scale/shift = silu(emb) @ emb_W + emb_b, split in halves.
// ---------------------------------------------------------------------------
__global__ void emb_kernel(const float* __restrict__ emb,
                           const float* __restrict__ W,
                           const float* __restrict__ b,
                           float* __restrict__ scale,
                           float* __restrict__ shift) {
    int t = threadIdx.x; // 256 threads, one output column each
    for (int bb = 0; bb < 4; ++bb) {
        float acc = b[t];
        for (int e = 0; e < 512; ++e) {
            float x = emb[bb * 512 + e];
            acc += silu_f(x) * W[e * 256 + t];
        }
        if (t < 128) scale[bb * 128 + t] = acc;
        else         shift[bb * 128 + (t - 128)] = acc;
    }
}

// ---------------------------------------------------------------------------
// Pack conv weights f32 [27][128cin][128cout] -> f16 WMMA B fragments.
// Fragment f = (k*4+kc)*8+nt holds B tile (K = kc*32..+31, N = nt*16..+15),
// laid out per the CDNA5 16-bit matrix VGPR layout (lane,halfidx).
// ---------------------------------------------------------------------------
__global__ void packw_kernel(const float* __restrict__ W,
                             _Float16* __restrict__ wf) {
    int u = blockIdx.x * 256 + threadIdx.x;
    if (u >= WFRAG_HALFS) return;
    int f = u >> 9;          // fragment id
    int r = u & 511;
    int L = r >> 4;          // lane 0..31
    int i = r & 15;          // half index within v16h
    int k  = f >> 5;
    int kc = (f >> 3) & 3;
    int nt = f & 7;
    int K    = kc * 32 + ((i & 8) << 1) + ((L & 16) >> 1) + (i & 7);
    int cout = nt * 16 + (L & 15);
    wf[u] = (_Float16)W[(k * 128 + K) * 128 + cout];
}

// ---------------------------------------------------------------------------
// Zero pad row N of both activation buffers (targets for invalid neighbors).
// ---------------------------------------------------------------------------
__global__ void zero_row_kernel(_Float16* __restrict__ h1,
                                _Float16* __restrict__ h2, int N) {
    int t = threadIdx.x; // 128
    h1[(size_t)N * 128 + t] = (_Float16)0.0f;
    h2[(size_t)N * 128 + t] = (_Float16)0.0f;
}

// ---------------------------------------------------------------------------
// h1 = silu(layernorm(feats, gamma, beta)) in f16. One block (128 thr) / row.
// ---------------------------------------------------------------------------
__global__ void ln_silu_kernel(const float* __restrict__ feats,
                               const float* __restrict__ g,
                               const float* __restrict__ b,
                               _Float16* __restrict__ h1, int N) {
    int row = blockIdx.x;
    int tid = threadIdx.x;
    __shared__ float r1[128], r2[128];
    float v = feats[row * 128 + tid];
    r1[tid] = v; r2[tid] = v * v;
    __syncthreads();
    for (int s = 64; s > 0; s >>= 1) {
        if (tid < s) { r1[tid] += r1[tid + s]; r2[tid] += r2[tid + s]; }
        __syncthreads();
    }
    float mu  = r1[0] * (1.0f / 128.0f);
    float var = r2[0] * (1.0f / 128.0f) - mu * mu;
    float y = (v - mu) * rsqrtf(var + 1e-6f) * g[tid] + b[tid];
    h1[row * 128 + tid] = (_Float16)silu_f(y);
}

// ---------------------------------------------------------------------------
// Core sparse-conv accumulator: one 16-row M-tile x 2 N-tiles per wave.
// Invalid neighbors are redirected to the all-zero row N: no masking VALU,
// no WMMA->VALU hazard nops, EXEC stays all-ones for WMMA.
// ---------------------------------------------------------------------------
__device__ __forceinline__ void spconv_acc(const _Float16* __restrict__ hin,
                                           const _Float16* __restrict__ wfrag,
                                           const int* __restrict__ sidx,
                                           int lane, int ntpair, int zrow,
                                           v8f& acc0, v8f& acc1) {
    const int klo  = (lane & 16) >> 1;  // 0 or 8 (per 16-bit A VGPR layout)
    const int mrow = lane & 15;
    const int nt0  = ntpair * 2;
    for (int k = 0; k < CCH; ++k) {
        int gi = sidx[mrow * 27 + k];
        if (__builtin_amdgcn_ballot_w32(gi >= 0) == 0u) continue; // uniform skip
        int safe = gi < 0 ? zrow : gi;
        const _Float16* __restrict__ ap = hin + (size_t)safe * 128 + klo;
#pragma unroll
        for (int kc = 0; kc < 4; ++kc) {
            union { v16h v; v8h h[2]; } a;
            a.h[0] = *(const v8h*)(ap + kc * 32);
            a.h[1] = *(const v8h*)(ap + kc * 32 + 16);
            const _Float16* wp = wfrag + (((k * 4 + kc) * 8 + nt0) * 512) + lane * 16;
            v16h b0 = *(const v16h*)(wp);
            v16h b1 = *(const v16h*)(wp + 512);
            acc0 = __builtin_amdgcn_wmma_f32_16x16x32_f16(
                false, a.v, false, b0, (short)0, acc0, false, false);
            acc1 = __builtin_amdgcn_wmma_f32_16x16x32_f16(
                false, a.v, false, b1, (short)0, acc1, false, false);
        }
    }
}

// ---------------------------------------------------------------------------
// conv1 + fused (bias, layernorm, FiLM, SiLU) -> h2 (f16)
// 256 threads = 8 waves: wave>>2 selects M-tile, wave&3 selects N-tile pair.
// ---------------------------------------------------------------------------
__global__ __launch_bounds__(256)
void conv1_kernel(const _Float16* __restrict__ h1,
                  const _Float16* __restrict__ wfrag,
                  const int* __restrict__ nbr,
                  const float* __restrict__ bias,
                  const float* __restrict__ scale,
                  const float* __restrict__ shift,
                  const int* __restrict__ batch_idx,
                  _Float16* __restrict__ h2, int N) {
    const int tid    = threadIdx.x;
    const int lane   = tid & 31;
    const int wave   = tid >> 5;
    const int mt     = wave >> 2;    // 0 or 1
    const int ntpair = wave & 3;     // 0..3
    const int m0     = blockIdx.x * MT;

    __shared__ int   sidx[MT * 27];
    __shared__ float outb[MT * 128];
    __shared__ float mus[MT], rss[MT];
    __shared__ int   brow[MT];

    for (int j = tid; j < MT * 27; j += 256) {
        int r = m0 + j / 27;
        sidx[j] = (r < N) ? nbr[r * 27 + (j % 27)] : -1;
    }
    __syncthreads();

    v8f acc0 = {}, acc1 = {};
    spconv_acc(h1, wfrag, sidx + mt * 16 * 27, lane, ntpair, N, acc0, acc1);

    // D layout: n = lane&15, m = r + 8*(lane>>4)
    {
        int rbase = mt * 16 + (lane >> 4) * 8;
        int c0 = ntpair * 32 + (lane & 15);
        int c1 = c0 + 16;
#pragma unroll
        for (int r = 0; r < 8; ++r) {
            outb[(rbase + r) * 128 + c0] = acc0[r];
            outb[(rbase + r) * 128 + c1] = acc1[r];
        }
    }
    __syncthreads();

    if (tid < MT) {
        float s = 0.f, s2 = 0.f;
        for (int c = 0; c < 128; ++c) {
            float v = outb[tid * 128 + c] + bias[c];
            s += v; s2 += v * v;
        }
        float mu  = s * (1.0f / 128.0f);
        float var = s2 * (1.0f / 128.0f) - mu * mu;
        mus[tid] = mu;
        rss[tid] = rsqrtf(var + 1e-6f);
        int rg = m0 + tid;
        brow[tid] = (rg < N) ? batch_idx[rg] : 0;
    }
    __syncthreads();

#pragma unroll
    for (int j = 0; j < 16; ++j) {
        int e = tid * 16 + j;          // 4096 elems / 256 threads
        int r = e >> 7, c = e & 127;
        float v = outb[e] + bias[c];
        float y = (v - mus[r]) * rss[r];
        int bb = brow[r];
        y = y * (1.0f + scale[bb * 128 + c]) + shift[bb * 128 + c];
        int rg = m0 + r;
        if (rg < N) h2[(size_t)rg * 128 + c] = (_Float16)silu_f(y);
    }
}

// ---------------------------------------------------------------------------
// conv2 + bias + residual -> d_out (f32), stored straight from D fragments.
// ---------------------------------------------------------------------------
__global__ __launch_bounds__(256)
void conv2_kernel(const _Float16* __restrict__ h2,
                  const _Float16* __restrict__ wfrag,
                  const int* __restrict__ nbr,
                  const float* __restrict__ bias,
                  const float* __restrict__ feats,
                  float* __restrict__ out, int N) {
    const int tid    = threadIdx.x;
    const int lane   = tid & 31;
    const int wave   = tid >> 5;
    const int mt     = wave >> 2;
    const int ntpair = wave & 3;
    const int m0     = blockIdx.x * MT;

    __shared__ int sidx[MT * 27];
    for (int j = tid; j < MT * 27; j += 256) {
        int r = m0 + j / 27;
        sidx[j] = (r < N) ? nbr[r * 27 + (j % 27)] : -1;
    }
    __syncthreads();

    v8f acc0 = {}, acc1 = {};
    spconv_acc(h2, wfrag, sidx + mt * 16 * 27, lane, ntpair, N, acc0, acc1);

    int rbase = mt * 16 + (lane >> 4) * 8;
    int c0 = ntpair * 32 + (lane & 15);
    int c1 = c0 + 16;
#pragma unroll
    for (int r = 0; r < 8; ++r) {
        int rg = m0 + rbase + r;
        if (rg < N) {
            out[rg * 128 + c0] = acc0[r] + bias[c0] + feats[rg * 128 + c0];
            out[rg * 128 + c1] = acc1[r] + bias[c1] + feats[rg * 128 + c1];
        }
    }
}

// ---------------------------------------------------------------------------
extern "C" void kernel_launch(void* const* d_in, const int* in_sizes, int n_in,
                              void* d_out, int out_size, void* d_ws, size_t ws_size,
                              hipStream_t stream) {
    const float* feats = (const float*)d_in[0];
    const float* emb   = (const float*)d_in[1];
    const float* g1    = (const float*)d_in[2];
    const float* be1   = (const float*)d_in[3];
    const float* W1    = (const float*)d_in[4];
    const float* b1    = (const float*)d_in[5];
    const float* W2    = (const float*)d_in[6];
    const float* b2    = (const float*)d_in[7];
    const float* embW  = (const float*)d_in[8];
    const float* embB  = (const float*)d_in[9];
    const int*   nbr   = (const int*)d_in[10];
    const int*   bidx  = (const int*)d_in[11];

    const int N = in_sizes[0] / 128;

    char* ws = (char*)d_ws;
    float*    d_scale = (float*)(ws);                    // 4*128 f32
    float*    d_shift = (float*)(ws + 2048);             // 4*128 f32
    _Float16* w1f     = (_Float16*)(ws + 4096);
    _Float16* w2f     = (_Float16*)(ws + 4096 + (size_t)WFRAG_HALFS * 2);
    _Float16* h1      = (_Float16*)(ws + 4096 + (size_t)WFRAG_HALFS * 4);
    _Float16* h2      = h1 + (size_t)(N + 1) * 128;      // +1 zero row each

    emb_kernel<<<1, 256, 0, stream>>>(emb, embW, embB, d_scale, d_shift);

    int packBlocks = (WFRAG_HALFS + 255) / 256;
    packw_kernel<<<packBlocks, 256, 0, stream>>>(W1, w1f);
    packw_kernel<<<packBlocks, 256, 0, stream>>>(W2, w2f);

    zero_row_kernel<<<1, 128, 0, stream>>>(h1, h2, N);
    ln_silu_kernel<<<N, 128, 0, stream>>>(feats, g1, be1, h1, N);

    int mBlocks = (N + MT - 1) / MT;
    conv1_kernel<<<mBlocks, 256, 0, stream>>>(h1, w1f, nbr, b1, d_scale, d_shift,
                                              bidx, h2, N);
    conv2_kernel<<<mBlocks, 256, 0, stream>>>(h2, w2f, nbr, b2, feats,
                                              (float*)d_out, N);
}
